// MoE_42958262895126
// MI455X (gfx1250) — matrix operand order repaired and checked by reading, not compiled, not run on hardware
//
#include <hip/hip_runtime.h>
#include <hip/hip_bf16.h>
#include <math.h>

typedef __attribute__((ext_vector_type(16))) __bf16 v16bf;
typedef __attribute__((ext_vector_type(8)))  float  v8f;

#define D     1024
#define NEXP  64
#define ES    128
#define TOPK  4
#define NTOK  8192   // B*S = 4*2048

#define XS 1032      // x tile row stride (shorts): 16B aligned, rows step 4 banks
#define HS 136       // H tile row stride (shorts): 16B aligned

__device__ __forceinline__ unsigned short f2bf(float f) {
    unsigned u = __builtin_bit_cast(unsigned, f);
    u += 0x7FFFu + ((u >> 16) & 1u);     // round-to-nearest-even
    return (unsigned short)(u >> 16);
}
__device__ __forceinline__ unsigned pack2(float a, float b) {
    return (unsigned)f2bf(a) | ((unsigned)f2bf(b) << 16);
}

struct Frag { uint4 lo; uint4 hi; };
__device__ __forceinline__ v16bf frag_cast(uint4 lo, uint4 hi) {
    Frag f{lo, hi};
    return __builtin_bit_cast(v16bf, f);
}

__global__ void zero_kernel(float* out, int n, int* counts) {
    int i = blockIdx.x * blockDim.x + threadIdx.x;
    if (i < n) out[i] = 0.0f;
    if (i < NEXP) counts[i] = 0;
}

// ---- one-time f32 -> bf16 conversion of keys into B-fragment layout ----
// kfrag index: ((((e*32 + kchunk)*8 + ntile)*32 + lane)*16 + j), lane = hf*16+n, j = k&15
__global__ void kconv_kernel(const float* __restrict__ keys,
                             unsigned short* __restrict__ kfrag) {
    int tid = blockIdx.x * blockDim.x + threadIdx.x;
    if (tid >= NEXP * (D / 4) * ES) return;
    int c  = tid & (ES - 1);          // 0..127
    int k4 = (tid >> 7) & (D / 4 - 1); // 0..255 (k = 4*k4)
    int e  = tid >> 15;
    const float* src = keys + ((size_t)e * D + (size_t)k4 * 4) * ES + c;
    float f0 = src[0], f1 = src[ES], f2 = src[2 * ES], f3 = src[3 * ES];
    int kch = k4 >> 3;
    int hf  = (k4 >> 2) & 1;
    int j0  = (k4 & 3) * 4;
    int nt  = c >> 4, n = c & 15;
    unsigned short* dst = kfrag +
        ((((size_t)(e * 32 + kch) * 8 + nt) * 32 + hf * 16 + n) * 16 + j0);
    ((unsigned*)dst)[0] = pack2(f0, f1);
    ((unsigned*)dst)[1] = pack2(f2, f3);
}

// vfrag index: ((((e*4 + kchunk)*64 + ntile)*32 + lane)*16 + j)
__global__ void vconv_kernel(const float* __restrict__ values,
                             unsigned short* __restrict__ vfrag) {
    int tid = blockIdx.x * blockDim.x + threadIdx.x;
    if (tid >= NEXP * (ES / 4) * D) return;
    int c  = tid & (D - 1);            // 0..1023
    int k4 = (tid >> 10) & (ES / 4 - 1); // 0..31 (k = 4*k4)
    int e  = tid >> 15;
    const float* src = values + ((size_t)e * ES + (size_t)k4 * 4) * D + c;
    float f0 = src[0], f1 = src[D], f2 = src[2 * D], f3 = src[3 * D];
    int kch = k4 >> 3;
    int hf  = (k4 >> 2) & 1;
    int j0  = (k4 & 3) * 4;
    int nt  = c >> 4, n = c & 15;
    unsigned short* dst = vfrag +
        ((((size_t)(e * 4 + kch) * 64 + nt) * 32 + hf * 16 + n) * 16 + j0);
    ((unsigned*)dst)[0] = pack2(f0, f1);
    ((unsigned*)dst)[1] = pack2(f2, f3);
}

// One 64-thread block per token: logits in f32, sigmoid, top-4 (exact selection)
__global__ void gating_kernel(const float* __restrict__ x,
                              const float* __restrict__ esel,
                              float* __restrict__ sel_val,
                              int* __restrict__ sel_idx,
                              int* __restrict__ counts) {
    __shared__ float xrow[D];
    __shared__ float sg[NEXP];
    int n = blockIdx.x;
    int tid = threadIdx.x;                     // 64 threads
    const float4* xr = (const float4*)(x + (size_t)n * D);
    for (int i = tid; i < D / 4; i += 64) ((float4*)xrow)[i] = xr[i];
    __syncthreads();
    const float4* w  = (const float4*)(esel + (size_t)tid * D);
    const float4* xl = (const float4*)xrow;
    float acc = 0.f;
    for (int i = 0; i < D / 4; ++i) {
        float4 a = xl[i], b = w[i];
        acc += a.x * b.x + a.y * b.y + a.z * b.z + a.w * b.w;
    }
    sg[tid] = 1.0f / (1.0f + __expf(-acc));
    __syncthreads();
    if (tid == 0) {
        float v[NEXP];
        for (int i = 0; i < NEXP; ++i) v[i] = sg[i];
        for (int h = 0; h < TOPK; ++h) {
            float best = -1.0f; int bi = 0;
            for (int i = 0; i < NEXP; ++i) if (v[i] > best) { best = v[i]; bi = i; }
            sel_val[n * TOPK + h] = best;
            sel_idx[n * TOPK + h] = bi;
            v[bi] = -2.0f;
            atomicAdd(&counts[bi], 1);
        }
    }
}

__global__ void scan_kernel(const int* __restrict__ counts,
                            int* __restrict__ offsets, int* __restrict__ cursors) {
    if (threadIdx.x == 0) {
        int acc = 0;
        for (int e = 0; e < NEXP; ++e) { offsets[e] = acc; acc += counts[e]; cursors[e] = 0; }
        offsets[NEXP] = acc;
    }
}

__global__ void fill_kernel(const float* __restrict__ sel_val,
                            const int* __restrict__ sel_idx,
                            const int* __restrict__ offsets, int* __restrict__ cursors,
                            int* __restrict__ pair_token, float* __restrict__ pair_gate) {
    int n = blockIdx.x * blockDim.x + threadIdx.x;
    if (n >= NTOK) return;
    for (int h = 0; h < TOPK; ++h) {
        int e = sel_idx[n * TOPK + h];
        int pos = atomicAdd(&cursors[e], 1);
        int slot = offsets[e] + pos;
        pair_token[slot] = n;
        pair_gate[slot]  = sel_val[n * TOPK + h];
    }
}

// Grouped GEMM over (expert, 16-token tile). 4 waves (128 threads).
// B operands come pre-converted/pre-swizzled from global (L2-resident bf16),
// fetched as two contiguous b128 per lane. A operands via LDS, two aligned
// ds_load_b128 per lane. Hot loop = loads + v_wmma only.
__global__ void __launch_bounds__(128)
moe_expert_kernel(const float* __restrict__ x,
                  const unsigned short* __restrict__ kfrag,
                  const unsigned short* __restrict__ vfrag,
                  const int* __restrict__ offsets,
                  const int* __restrict__ pair_token,
                  const float* __restrict__ pair_gate,
                  float* __restrict__ out) {
    __shared__ __align__(16) unsigned char smem[16 * XS * 2 + 16 * HS * 2 + 128];
    unsigned short* xs  = (unsigned short*)smem;                        // 33024 B
    unsigned short* hs  = (unsigned short*)(smem + 16 * XS * 2);        // 4352 B
    float*          gsh = (float*)(smem + 16 * XS * 2 + 16 * HS * 2);   // 16 gates
    int*            tsh = (int*)  (smem + 16 * XS * 2 + 16 * HS * 2 + 64);

    int e = blockIdx.x;
    int t = blockIdx.y;
    int off = offsets[e];
    int cnt = offsets[e + 1] - off;
    if (t * 16 >= cnt) return;
    int valid = min(16, cnt - t * 16);

    int tid  = threadIdx.x;
    int wid  = tid >> 5;
    int lane = tid & 31;
    int hf   = lane >> 4;
    int ln   = lane & 15;

    if (tid < 16) {
        if (tid < valid) {
            int slot = off + t * 16 + tid;
            gsh[tid] = pair_gate[slot];
            tsh[tid] = pair_token[slot];
        } else { gsh[tid] = 0.0f; tsh[tid] = 0; }
    }
    // Gather 16 token rows of x -> bf16 LDS (8 threads per row, dword stores)
    {
        int row = tid >> 3, l8 = tid & 7;
        unsigned* xd = (unsigned*)(xs + row * XS);
        if (row < valid) {
            int tk = pair_token[off + t * 16 + row];
            const float4* xr = (const float4*)(x + (size_t)tk * D);
            for (int i = l8; i < D / 4; i += 8) {
                float4 f = xr[i];
                xd[i * 2 + 0] = pack2(f.x, f.y);
                xd[i * 2 + 1] = pack2(f.z, f.w);
            }
        } else {
            for (int i = l8; i < D / 4; i += 8) { xd[i * 2 + 0] = 0; xd[i * 2 + 1] = 0; }
        }
    }
    __syncthreads();

    // ---- Stage A: H[16x128] = relu(X @ keys[e]) * gate, K = 1024 ----
    v8f acc0 = {}; v8f acc1 = {};
    int nt0 = wid * 2, nt1 = wid * 2 + 1;               // 8 N-tiles over 4 waves
    const unsigned short* kf = kfrag + (size_t)e * (32 * 8 * 32 * 16);
    for (int kch = 0; kch < 32; ++kch) {
        const unsigned short* ab = xs + ln * XS + kch * 32 + 16 * hf;
        v16bf a = frag_cast(*(const uint4*)ab, *(const uint4*)(ab + 8));
        const uint4* b0p = (const uint4*)(kf + ((size_t)(kch * 8 + nt0) * 32 + lane) * 16);
        const uint4* b1p = (const uint4*)(kf + ((size_t)(kch * 8 + nt1) * 32 + lane) * 16);
        v16bf b0 = frag_cast(b0p[0], b0p[1]);
        v16bf b1 = frag_cast(b1p[0], b1p[1]);
        acc0 = __builtin_amdgcn_wmma_f32_16x16x32_bf16(false, a, false, b0, (short)0, acc0, false, false);
        acc1 = __builtin_amdgcn_wmma_f32_16x16x32_bf16(false, a, false, b1, (short)0, acc1, false, false);
    }
    // relu * gate -> bf16 H tile. C layout: element(m,n): lane = n + 16*(m>=8), vgpr r = m%8
    #pragma unroll
    for (int r = 0; r < 8; ++r) {
        int m = r + 8 * hf;
        float g  = gsh[m];
        hs[m * HS + nt0 * 16 + ln] = f2bf(fmaxf(acc0[r], 0.0f) * g);
        hs[m * HS + nt1 * 16 + ln] = f2bf(fmaxf(acc1[r], 0.0f) * g);
    }
    __syncthreads();

    // ---- Stage B: OUT[16x1024] = H @ values[e], K = 128, 4 column passes ----
    const unsigned short* vf = vfrag + (size_t)e * (4 * 64 * 32 * 16);
    for (int p = 0; p < 4; ++p) {
        v8f acc[4] = { {}, {}, {}, {} };
        for (int kch = 0; kch < 4; ++kch) {
            const unsigned short* ab = hs + ln * HS + kch * 32 + 16 * hf;
            v16bf a = frag_cast(*(const uint4*)ab, *(const uint4*)(ab + 8));
            #pragma unroll
            for (int q = 0; q < 4; ++q) {
                int nt = p * 16 + wid * 4 + q;          // 0..63
                const uint4* bp = (const uint4*)(vf + ((size_t)(kch * 64 + nt) * 32 + lane) * 16);
                v16bf b = frag_cast(bp[0], bp[1]);
                acc[q] = __builtin_amdgcn_wmma_f32_16x16x32_bf16(false, a, false, b, (short)0, acc[q], false, false);
            }
        }
        #pragma unroll
        for (int q = 0; q < 4; ++q) {
            int col = p * 256 + wid * 64 + q * 16 + ln;
            #pragma unroll
            for (int r = 0; r < 8; ++r) {
                int m = r + 8 * hf;
                atomicAdd(&out[(size_t)tsh[m] * D + col], acc[q][r]);
            }
        }
    }
}

extern "C" void kernel_launch(void* const* d_in, const int* in_sizes, int n_in,
                              void* d_out, int out_size, void* d_ws, size_t ws_size,
                              hipStream_t stream) {
    const float* x      = (const float*)d_in[0];
    const float* esel   = (const float*)d_in[1];
    const float* keys   = (const float*)d_in[2];
    const float* values = (const float*)d_in[3];
    float* out = (float*)d_out;

    // workspace: 0.53 MB control + 16 MB kfrag + 16 MB vfrag  (~32.5 MB used, 33 MB span)
    unsigned char* ws = (unsigned char*)d_ws;
    float* sel_val  = (float*)(ws + 0x00000);
    int*   sel_idx  = (int*)  (ws + 0x20000);
    int*   pair_tok = (int*)  (ws + 0x40000);
    float* pair_gat = (float*)(ws + 0x60000);
    int*   counts   = (int*)  (ws + 0x80000);
    int*   offsets  = (int*)  (ws + 0x80200);
    int*   cursors  = (int*)  (ws + 0x80400);
    unsigned short* kfrag = (unsigned short*)(ws + 0x0100000);  // 16 MB
    unsigned short* vfrag = (unsigned short*)(ws + 0x1100000);  // 16 MB

    zero_kernel<<<(out_size + 255) / 256, 256, 0, stream>>>(out, out_size, counts);
    kconv_kernel<<<(NEXP * (D / 4) * ES + 255) / 256, 256, 0, stream>>>(keys, kfrag);
    vconv_kernel<<<(NEXP * (ES / 4) * D + 255) / 256, 256, 0, stream>>>(values, vfrag);
    gating_kernel<<<NTOK, 64, 0, stream>>>(x, esel, sel_val, sel_idx, counts);
    scan_kernel<<<1, 32, 0, stream>>>(counts, offsets, cursors);
    fill_kernel<<<(NTOK + 255) / 256, 256, 0, stream>>>(sel_val, sel_idx, offsets, cursors,
                                                        pair_tok, pair_gat);
    dim3 grid(NEXP, NTOK / 16);   // worst case: one expert takes every token
    moe_expert_kernel<<<grid, 128, 0, stream>>>(x, kfrag, vfrag, offsets, pair_tok, pair_gat, out);
}
